// PatchGCN_43782896615726
// MI455X (gfx1250) — matrix operand order repaired and checked by reading, not compiled, not run on hardware
//
#include <hip/hip_runtime.h>
#include <hip/hip_bf16.h>
#include <math.h>

// ---------------------------------------------------------------------------
// PatchGCN forward for MI455X (gfx1250, wave32).
// Dense GEMMs: v_wmma_f32_16x16x32_bf16, fragments staged in LDS in native
// WMMA lane order so each fragment is exactly 2x ds_load_b128.
// Graph softmax aggregation: global float atomics (values strictly positive).
// ---------------------------------------------------------------------------

typedef __attribute__((ext_vector_type(8)))  float          v8f;
typedef __attribute__((ext_vector_type(16))) __bf16         v16bf;
typedef __attribute__((ext_vector_type(8)))  unsigned short ush8;

union Frag { v16bf v; ush8 u[2]; };

__device__ __forceinline__ unsigned short f2bf(float f) {
    unsigned u = __float_as_uint(f);
    unsigned r = 0x7FFFu + ((u >> 16) & 1u);          // round-to-nearest-even
    return (unsigned short)((u + r) >> 16);
}

// ---------------- weight f32 -> bf16 conversion ----------------------------
__global__ void k_f32_to_bf16(const float* __restrict__ s,
                              unsigned short* __restrict__ d, int n) {
    int i = blockIdx.x * blockDim.x + threadIdx.x;
    if (i < n) d[i] = f2bf(s[i]);
}

// ---------------- generic zero ---------------------------------------------
__global__ void k_zero(float* __restrict__ p, int n) {
    int i = blockIdx.x * blockDim.x + threadIdx.x;
    if (i < n) p[i] = 0.0f;
}

// ---------------- WMMA GEMM: C = act(A[M,K] @ Bbf16[K,Nout] + bias) --------
// block = 128 threads (4 waves). Each wave computes a 16x64 output tile.
// LDS holds tiles pre-swizzled into WMMA fragment order:
//   lAf[wave][lane][16 halfs]   A 16x32 sub-tile for that wave
//   lBf[tcol][lane][16 halfs]   B 32x16 sub-tile for column group tcol
// mode: 0=linear 1=relu 2=tanh 3=sigmoid()*R 4=linear+R
__global__ __launch_bounds__(128)
void k_gemm_wmma(const float* __restrict__ A, int lda,
                 const unsigned short* __restrict__ B, int ldb,
                 const float* __restrict__ bias,
                 float* __restrict__ C, int ldc,
                 const float* __restrict__ R, int ldr,
                 int M, int K, int mode)
{
    __shared__ unsigned short lAf[4][32][16];   // 4 KB, rows 32B aligned
    __shared__ unsigned short lBf[4][32][16];   // 4 KB

    const int tid  = threadIdx.x;
    const int wave = tid >> 5;
    const int lane = tid & 31;
    const int rowBase = blockIdx.x * 64;
    const int colBase = blockIdx.y * 64;

    v8f acc[4];
    #pragma unroll
    for (int t = 0; t < 4; ++t)
        #pragma unroll
        for (int r = 0; r < 8; ++r) acc[t][r] = 0.0f;

    for (int k0 = 0; k0 < K; k0 += 32) {
        // ---- stage A tile: 64 rows x 32 k (2 threads / row), fragment order
        {
            int row  = tid >> 1;
            int cb   = (tid & 1) * 16;
            int grow = rowBase + row;
            const float* ap = A + (size_t)grow * lda + k0 + cb;
            int w = row >> 4, m = row & 15;
            float vbuf[16];
            if (grow < M) {                       // single per-thread branch
                #pragma unroll
                for (int j = 0; j < 16; ++j) vbuf[j] = ap[j];
            } else {
                #pragma unroll
                for (int j = 0; j < 16; ++j) vbuf[j] = 0.0f;
            }
            #pragma unroll
            for (int j = 0; j < 16; ++j) {
                int k = cb + j;
                int lslot = m + (((k >> 3) & 1) << 4);     // lane holding (m,k)
                int idx   = (k & 7) + ((k & 16) ? 8 : 0);  // half index
                lAf[w][lslot][idx] = f2bf(vbuf[j]);
            }
        }
        // ---- stage B tile: 32 k x 64 cols (2 threads / k-row), fragment order
        {
            int kr = tid >> 1;
            int cb = (tid & 1) * 32;
            const ush8* bp = (const ush8*)(B + (size_t)(k0 + kr) * ldb + colBase + cb);
            unsigned short tmp[32];
            #pragma unroll
            for (int q = 0; q < 4; ++q) *(ush8*)&tmp[q * 8] = bp[q];
            int slotAdd = ((kr >> 3) & 1) << 4;
            int idx     = (kr & 7) + ((kr & 16) ? 8 : 0);
            #pragma unroll
            for (int j = 0; j < 32; ++j) {
                int c = cb + j;                            // 0..63
                lBf[c >> 4][slotAdd + (c & 15)][idx] = tmp[j];
            }
        }
        __syncthreads();

        // ---- fragments: 2x ds_load_b128 each
        Frag af;
        {
            const ush8* p = (const ush8*)&lAf[wave][lane][0];
            af.u[0] = p[0]; af.u[1] = p[1];
        }
        #pragma unroll
        for (int t = 0; t < 4; ++t) {
            Frag bf;
            const ush8* p = (const ush8*)&lBf[t][lane][0];
            bf.u[0] = p[0]; bf.u[1] = p[1];
            acc[t] = __builtin_amdgcn_wmma_f32_16x16x32_bf16(
                         false, af.v, false, bf.v, (short)0, acc[t], false, false);
        }
        __syncthreads();
    }

    // ---- epilogue (C/D layout: VGPR r -> M = r + (lane<16?0:8), N = lane&15)
    const int moff = (lane < 16) ? 0 : 8;
    const int nlan = lane & 15;
    #pragma unroll
    for (int t = 0; t < 4; ++t) {
        #pragma unroll
        for (int r = 0; r < 8; ++r) {
            int grow = rowBase + wave * 16 + moff + r;
            if (grow >= M) continue;
            int gcol = colBase + t * 16 + nlan;
            float v = acc[t][r] + bias[gcol];
            if      (mode == 1) v = fmaxf(v, 0.0f);
            else if (mode == 2) v = tanhf(v);
            else if (mode == 3) { v = 1.0f / (1.0f + expf(-v));
                                  v *= R[(size_t)grow * ldr + gcol]; }
            else if (mode == 4) { v += R[(size_t)grow * ldr + gcol]; }
            C[(size_t)grow * ldc + gcol] = v;
        }
    }
}

// ---------------- LayerNorm + ReLU, one wave per row -----------------------
template <int D>
__global__ void k_ln_relu(const float* __restrict__ X, float* __restrict__ Y,
                          const float* __restrict__ g, const float* __restrict__ b,
                          int N)
{
    const int wave = threadIdx.x >> 5;
    const int lane = threadIdx.x & 31;
    const int n = blockIdx.x * (blockDim.x >> 5) + wave;
    if (n >= N) return;
    const float* x = X + (size_t)n * D;
    constexpr int PER = D / 32;
    float v[PER];
    float s = 0.0f, s2 = 0.0f;
    #pragma unroll
    for (int i = 0; i < PER; ++i) {
        v[i] = x[lane + 32 * i];
        s += v[i]; s2 += v[i] * v[i];
    }
    #pragma unroll
    for (int o = 16; o > 0; o >>= 1) {
        s  += __shfl_xor(s,  o, 32);
        s2 += __shfl_xor(s2, o, 32);
    }
    float mu   = s / D;
    float var  = s2 / D - mu * mu;
    float rstd = rsqrtf(var + 1e-5f);
    float* y = Y + (size_t)n * D;
    #pragma unroll
    for (int i = 0; i < PER; ++i) {
        int c = lane + 32 * i;
        y[c] = fmaxf((v[i] - mu) * rstd * g[c] + b[c], 0.0f);
    }
}

// ---------------- copy 64-wide block into x_ (ld 256) ----------------------
__global__ void k_copy_cols(const float* __restrict__ src, float* __restrict__ dst,
                            int rows) {
    int i = blockIdx.x * blockDim.x + threadIdx.x;
    int r = i >> 6, c = i & 63;
    if (r < rows) dst[(size_t)r * 256 + c] = src[(size_t)r * 64 + c];
}

// ---------------- edge pass 1: segment max (positive floats as uints) ------
__global__ void k_edge_pass1(const float* __restrict__ hv1,
                             const int* __restrict__ src, const int* __restrict__ dst,
                             const float* __restrict__ efw, const float* __restrict__ efb,
                             const float* __restrict__ beta,
                             unsigned* __restrict__ mxu, int E)
{
    int t = blockIdx.x * blockDim.x + threadIdx.x;
    int e = t >> 2;
    if (e >= E) return;
    int c0 = (t & 3) * 16;
    int sn = src[e], dn = dst[e];
    float bet = beta[0];
    const float* hp = hv1 + (size_t)sn * 64 + c0;
    unsigned* mp = mxu + (size_t)dn * 64 + c0;
    #pragma unroll
    for (int j = 0; j < 16; ++j) {
        float he = fmaxf(efw[c0 + j] + efb[c0 + j], 0.0f);
        float m  = fmaxf(hp[j] + he, 0.0f) + 1e-7f;   // m > 0 always
        atomicMax(&mp[j], __float_as_uint(m * bet));  // positive => uint order ok
    }
}

// ---------------- edge pass 2: sum(e) and sum(m*e) -------------------------
// agg = sum(m*e)/sum(e) since s[dst] is constant within a segment/channel.
__global__ void k_edge_pass2(const float* __restrict__ hv1,
                             const int* __restrict__ src, const int* __restrict__ dst,
                             const float* __restrict__ efw, const float* __restrict__ efb,
                             const float* __restrict__ beta,
                             const unsigned* __restrict__ mxu,
                             float* __restrict__ ssum, float* __restrict__ tsum, int E)
{
    int t = blockIdx.x * blockDim.x + threadIdx.x;
    int e = t >> 2;
    if (e >= E) return;
    int c0 = (t & 3) * 16;
    int sn = src[e], dn = dst[e];
    float bet = beta[0];
    const float* hp = hv1 + (size_t)sn * 64 + c0;
    const unsigned* mp = mxu + (size_t)dn * 64 + c0;
    float* sp = ssum + (size_t)dn * 64 + c0;
    float* tp = tsum + (size_t)dn * 64 + c0;
    #pragma unroll
    for (int j = 0; j < 16; ++j) {
        float he = fmaxf(efw[c0 + j] + efb[c0 + j], 0.0f);
        float m  = fmaxf(hp[j] + he, 0.0f) + 1e-7f;
        float ex = expf(m * bet - __uint_as_float(mp[j]));
        atomicAdd(&sp[j], ex);
        atomicAdd(&tp[j], m * ex);
    }
}

// ---------------- feats = hv1 + agg ----------------------------------------
__global__ void k_combine(const float* __restrict__ hv1, const float* __restrict__ ssum,
                          const float* __restrict__ tsum, float* __restrict__ feats, int n)
{
    int i = blockIdx.x * blockDim.x + threadIdx.x;
    if (i < n) {
        float s = ssum[i];
        feats[i] = hv1[i] + ((s > 0.0f) ? tsum[i] / s : 0.0f);
    }
}

// ---------------- gated attention score + global max -----------------------
__global__ void k_gate(const float* __restrict__ ab, const float* __restrict__ cw,
                       const float* __restrict__ cb, float* __restrict__ gate,
                       unsigned* __restrict__ gmax, int N)
{
    const int wave = threadIdx.x >> 5;
    const int lane = threadIdx.x & 31;
    const int n = blockIdx.x * (blockDim.x >> 5) + wave;
    if (n >= N) return;
    float acc = 0.0f;
    #pragma unroll
    for (int i = 0; i < 8; ++i) {
        int c = lane + 32 * i;
        acc += ab[(size_t)n * 256 + c] * cw[c];
    }
    #pragma unroll
    for (int o = 16; o > 0; o >>= 1) acc += __shfl_xor(acc, o, 32);
    if (lane == 0) {
        float g = acc + cb[0];
        gate[n] = g;
        unsigned ub  = __float_as_uint(g);
        unsigned key = (ub & 0x80000000u) ? ~ub : (ub | 0x80000000u); // orderable
        atomicMax(gmax, key);
    }
}

// ---------------- softmax-weighted pooling ---------------------------------
__global__ void k_pool(const float* __restrict__ gate, const unsigned* __restrict__ gmaxu,
                       const float* __restrict__ hp, float* __restrict__ hg,
                       float* __restrict__ sumw, int N)
{
    __shared__ float w[256];
    __shared__ float bsum;
    const int tid  = threadIdx.x;
    const int base = blockIdx.x * 256;
    if (tid == 0) bsum = 0.0f;
    __syncthreads();
    unsigned key = *gmaxu;
    float mx = __uint_as_float((key & 0x80000000u) ? (key & 0x7FFFFFFFu) : ~key);
    int n = base + tid;
    float wv = (n < N) ? expf(gate[n] - mx) : 0.0f;
    w[tid] = wv;
    atomicAdd(&bsum, wv);
    __syncthreads();
    float acc = 0.0f;
    int lim = (N - base < 256) ? (N - base) : 256;
    for (int i = 0; i < lim; ++i)
        acc += w[i] * hp[(size_t)(base + i) * 256 + tid];
    atomicAdd(&hg[tid], acc);
    if (tid == 0) atomicAdd(sumw, bsum);
}

// ---------------- head: rho + relu, classifier -----------------------------
__global__ void k_final(const float* __restrict__ hg, const float* __restrict__ sumw,
                        const float* __restrict__ rho_w, const float* __restrict__ rho_b,
                        const float* __restrict__ cls_w, const float* __restrict__ cls_b,
                        float* __restrict__ out)
{
    __shared__ float h[256];
    __shared__ float of[64];
    const int tid = threadIdx.x;
    float inv = 1.0f / (*sumw);
    h[tid] = hg[tid] * inv;
    __syncthreads();
    if (tid < 64) {
        float a = rho_b[tid];
        for (int c = 0; c < 256; ++c) a += h[c] * rho_w[c * 64 + tid];
        a = fmaxf(a, 0.0f);
        of[tid] = a;
        out[2 + tid] = a;          // out_feat
    }
    __syncthreads();
    if (tid < 2) {
        float a = cls_b[tid];
        for (int k = 0; k < 64; ++k) a += of[k] * cls_w[k * 2 + tid];
        out[tid] = a;              // out
    }
}

// ===========================================================================
extern "C" void kernel_launch(void* const* d_in, const int* in_sizes, int n_in,
                              void* d_out, int out_size, void* d_ws, size_t ws_size,
                              hipStream_t stream)
{
    (void)in_sizes; (void)n_in; (void)out_size; (void)ws_size;
    const int N = 50000, E = 800000, F = 1024, H = 64, NL = 3;

    // ---- input pointers (setup_inputs dict order, params flattened in
    //      insertion order) ----
    const float* node_feat = (const float*)d_in[0];
    const int*   src       = (const int*)d_in[1];
    const int*   dst       = (const int*)d_in[2];
    const float* n_fc_w    = (const float*)d_in[3];
    const float* n_fc_b    = (const float*)d_in[4];
    const float* e_fc_w    = (const float*)d_in[5];
    const float* e_fc_b    = (const float*)d_in[6];
    // layer l base = 7 + 9*l :
    //   +0 norm_g +1 norm_b +2 beta +3 mlp_w1 +4 mlp_b1 +5 ln_g +6 ln_b +7 mlp_w2 +8 mlp_b2
    const float* phi_w   = (const float*)d_in[34];
    const float* phi_b   = (const float*)d_in[35];
    const float* att_a_w = (const float*)d_in[36];
    const float* att_a_b = (const float*)d_in[37];
    const float* att_b_w = (const float*)d_in[38];
    const float* att_b_b = (const float*)d_in[39];
    const float* att_c_w = (const float*)d_in[40];
    const float* att_c_b = (const float*)d_in[41];
    const float* rho_w   = (const float*)d_in[42];
    const float* rho_b   = (const float*)d_in[43];
    const float* cls_w   = (const float*)d_in[44];
    const float* cls_b   = (const float*)d_in[45];

    // ---- workspace carve (floats) ----
    float* ws = (float*)d_ws;
    size_t off = 0;
    float* x_    = ws + off; off += (size_t)N * 256;   // concat feats; reused as a*b buf
    float* hv    = ws + off; off += (size_t)N * H;
    float* hv1   = ws + off; off += (size_t)N * H;
    float* feats = ws + off; off += (size_t)N * H;
    float* mx    = ws + off; off += (size_t)N * H;     // mx..ssum reused as tmp128
    float* ssum  = ws + off; off += (size_t)N * H;
    float* tsum  = ws + off; off += (size_t)N * H;
    float* gate  = ws + off; off += (size_t)N;
    float* scal  = ws + off; off += 260;               // [0]=gmax(u32) [1]=sumw [2..257]=hg
    off = (off + 3) & ~(size_t)3;                      // 16B-align bf16 arena
    unsigned short* bfw = (unsigned short*)(ws + off);
    float* tmp128 = mx;
    float* abuf   = x_;
    float* h_path = (float*)d_out + 66;                // d_out: [0:2]=out [2:66]=out_feat

    unsigned short* bf_n   = bfw;                       // 1024x64
    unsigned short* bf_w1  = bfw + 65536;               // 3 x 64x128
    unsigned short* bf_w2  = bfw + 65536 + 3 * 8192;    // 3 x 128x64
    unsigned short* bf_phi = bfw + 65536 + 6 * 8192;    // 256x256
    unsigned short* bf_a   = bf_phi + 65536;
    unsigned short* bf_b   = bf_a + 65536;

    auto cvt = [&](const float* s, unsigned short* d, int n) {
        k_f32_to_bf16<<<(n + 255) / 256, 256, 0, stream>>>(s, d, n);
    };
    auto gemm = [&](const float* A, int lda, const unsigned short* B, int ldb,
                    const float* bias, float* C, int ldc,
                    const float* R, int ldr, int M, int K, int Nout, int mode) {
        dim3 g((M + 63) / 64, Nout / 64);
        k_gemm_wmma<<<g, 128, 0, stream>>>(A, lda, B, ldb, bias, C, ldc, R, ldr, M, K, mode);
    };

    // ---- weights to bf16 ----
    cvt(n_fc_w, bf_n, F * H);
    for (int l = 0; l < NL; ++l) {
        cvt((const float*)d_in[7 + 9 * l + 3], bf_w1 + l * 8192, H * 2 * H);
        cvt((const float*)d_in[7 + 9 * l + 7], bf_w2 + l * 8192, 2 * H * H);
    }
    cvt(phi_w, bf_phi, 256 * 256);
    cvt(att_a_w, bf_a, 256 * 256);
    cvt(att_b_w, bf_b, 256 * 256);

    // ---- n_fc: hv = relu(node_feat @ W + b) ; memory-bound (205 MB @ 23.3 TB/s)
    gemm(node_feat, F, bf_n, H, n_fc_b, hv, H, nullptr, 0, N, F, H, /*relu*/1);
    k_copy_cols<<<(N * 64 + 255) / 256, 256, 0, stream>>>(hv, x_, N);

    // ---- 3 GENConv layers ----
    for (int l = 0; l < NL; ++l) {
        const int base = 7 + 9 * l;
        const float* norm_g = (const float*)d_in[base + 0];
        const float* norm_b = (const float*)d_in[base + 1];
        const float* beta   = (const float*)d_in[base + 2];
        const float* mlp_b1 = (const float*)d_in[base + 4];
        const float* ln_g   = (const float*)d_in[base + 5];
        const float* ln_b   = (const float*)d_in[base + 6];
        const float* mlp_b2 = (const float*)d_in[base + 8];

        k_ln_relu<64><<<(N + 7) / 8, 256, 0, stream>>>(hv, hv1, norm_g, norm_b, N);
        k_zero<<<(3 * N * H + 255) / 256, 256, 0, stream>>>(mx, 3 * N * H);
        int egrid = (E * 4 + 255) / 256;
        k_edge_pass1<<<egrid, 256, 0, stream>>>(hv1, src, dst, e_fc_w, e_fc_b, beta,
                                                (unsigned*)mx, E);
        k_edge_pass2<<<egrid, 256, 0, stream>>>(hv1, src, dst, e_fc_w, e_fc_b, beta,
                                                (const unsigned*)mx, ssum, tsum, E);
        k_combine<<<(N * H + 255) / 256, 256, 0, stream>>>(hv1, ssum, tsum, feats, N * H);
        // MLP: Linear(64->128) -> LN -> ReLU -> Linear(128->64), residual to hv
        gemm(feats, H, bf_w1 + l * 8192, 2 * H, mlp_b1, tmp128, 2 * H,
             nullptr, 0, N, H, 2 * H, /*linear*/0);
        k_ln_relu<128><<<(N + 7) / 8, 256, 0, stream>>>(tmp128, tmp128, ln_g, ln_b, N);
        gemm(tmp128, 2 * H, bf_w2 + l * 8192, H, mlp_b2, hv, H,
             hv, H, N, 2 * H, H, /*+residual*/4);
        k_copy_cols<<<(N * 64 + 255) / 256, 256, 0, stream>>>(hv, x_ + 64 * (l + 1), N);
    }

    // ---- h_path = relu(x_ @ phi) -> d_out ; gated attention ----
    gemm(x_, 256, bf_phi, 256, phi_b, h_path, 256, nullptr, 0, N, 256, 256, /*relu*/1);
    gemm(h_path, 256, bf_a, 256, att_a_b, abuf, 256, nullptr, 0, N, 256, 256, /*tanh*/2);
    gemm(h_path, 256, bf_b, 256, att_b_b, abuf, 256, abuf, 256, N, 256, 256, /*sig*mul*/3);

    k_zero<<<2, 256, 0, stream>>>(scal, 260);
    k_gate<<<(N + 7) / 8, 256, 0, stream>>>(abuf, att_c_w, att_c_b, gate,
                                            (unsigned*)scal, N);
    k_pool<<<(N + 255) / 256, 256, 0, stream>>>(gate, (const unsigned*)scal, h_path,
                                                scal + 2, scal + 1, N);
    k_final<<<1, 256, 0, stream>>>(scal + 2, scal + 1, rho_w, rho_b, cls_w, cls_b,
                                   (float*)d_out);
}
